// SpatialEncoder_70652212019229
// MI455X (gfx1250) — matrix operand order repaired and checked by reading, not compiled, not run on hardware
//
#include <hip/hip_runtime.h>
#include <hip/hip_bf16.h>

typedef __attribute__((ext_vector_type(16))) __bf16 v16bf;
typedef __attribute__((ext_vector_type(8)))  __bf16 v8bf;
typedef __attribute__((ext_vector_type(8)))  float  v8f;

// ---- problem constants ----
#define TPW    32          // tokens per workgroup (2 WMMA M-tiles)
#define FIN    64
#define NNODE  24
#define HDIM   64
#define NH     1536        // NNODE*HDIM
#define NT_WH  96          // N-tiles producing Wh
#define NT_ALL 99          // + 3 N-tiles producing [f1|f2] (48 cols)
#define WS_WH_ELEMS  (NT_WH*2*512)    // 98304 bf16
#define WS_ALL_ELEMS (NT_ALL*2*512)   // 101376 bf16

// -------------------------------------------------------------------------
// Kernel 0 (prep): build one fused, bf16, WMMA-B-swizzled weight matrix
//   cols [0,1536) : Wf = Wp.Wblocked           (Wh   = x.Wf + bp2)
//   cols [1536,1584): [wf1|wf2] = Wp.(W.a1|2)  (f1|f2 = x.wf + b1|b2)
// fragment order: frag = nt*2+ks ; element (lane,j) = M[k][n],
//   n = nt*16+(lane&15), k = ks*32 + j + 16*(lane>>4)
// plus fp32 fused biases bp2[1536] and b12[48].
// -------------------------------------------------------------------------
__global__ __launch_bounds__(256)
void gat_prep(const float* __restrict__ Wp, const float* __restrict__ bp,
              const float* __restrict__ W,  const float* __restrict__ avec,
              __bf16* __restrict__ wsB, float* __restrict__ wsBias,
              float* __restrict__ wsB12) {
    int idx = blockIdx.x * 256 + threadIdx.x;
    if (idx < WS_WH_ELEMS) {                       // Wf fragments
        int frag = idx >> 9, r = idx & 511;
        int lane = r >> 4, j = r & 15;
        int nt = frag >> 1, ks = frag & 1;
        int n = nt * 16 + (lane & 15);
        int k = ks * 32 + j + 16 * (lane >> 4);
        int nd = n >> 6, jc = n & 63;
        const float* wpRow = Wp + (size_t)k * NH + nd * 64;
        float acc = 0.f;
        for (int c = 0; c < 64; c++) acc += wpRow[c] * W[c * 64 + jc];
        wsB[idx] = (__bf16)acc;
    } else if (idx < WS_ALL_ELEMS) {               // [wf1|wf2] fragments
        int i2 = idx - WS_WH_ELEMS;
        int frag = i2 >> 9, r = i2 & 511;
        int lane = r >> 4, j = r & 15;
        int tt = frag >> 1, ks = frag & 1;
        int c = tt * 16 + (lane & 15);             // 0..47
        int k = ks * 32 + j + 16 * (lane >> 4);
        int ndd = (c < 24) ? c : c - 24;
        const float* av = (c < 24) ? avec : avec + HDIM;
        const float* wpRow = Wp + (size_t)k * NH + ndd * 64;
        float acc = 0.f;
        for (int cc = 0; cc < 64; cc++) {
            float wa = 0.f;
            for (int j2 = 0; j2 < 64; j2++) wa += W[cc * 64 + j2] * av[j2];
            acc += wpRow[cc] * wa;
        }
        wsB[idx] = (__bf16)acc;
    } else if (idx < WS_ALL_ELEMS + NH) {          // bp2 = bp.Wblocked
        int n = idx - WS_ALL_ELEMS;
        int nd = n >> 6, jc = n & 63;
        const float* bpRow = bp + nd * 64;
        float acc = 0.f;
        for (int c = 0; c < 64; c++) acc += bpRow[c] * W[c * 64 + jc];
        wsBias[n] = acc;
    } else if (idx < WS_ALL_ELEMS + NH + 48) {     // b1|b2 = bp.(W.a1|2)
        int c = idx - (WS_ALL_ELEMS + NH);
        int ndd = (c < 24) ? c : c - 24;
        const float* av = (c < 24) ? avec : avec + HDIM;
        const float* bpRow = bp + ndd * 64;
        float acc = 0.f;
        for (int cc = 0; cc < 64; cc++) {
            float wa = 0.f;
            for (int j2 = 0; j2 < 64; j2++) wa += W[cc * 64 + j2] * av[j2];
            acc += bpRow[cc] * wa;
        }
        wsB12[c] = acc;
    }
}

// -------------------------------------------------------------------------
// Kernel 1: fully fused GAT tile kernel. 1 WG = 32 tokens, 8 waves.
//   Phase 1 : Wh (token-minor LDS, b128 stores) + f1/f2, all via WMMA
//   Phase 2 : leaky-relu / adjacency mask / softmax (wave shuffles)
//   Phase 3 : h' = attn @ Wh, elu, mean over nodes -> out
// -------------------------------------------------------------------------
__global__ __launch_bounds__(256)
void gat_fused(const float* __restrict__ x, const int* __restrict__ adj,
               const __bf16* __restrict__ wsB, const float* __restrict__ wsBias,
               const float* __restrict__ wsB12, float* __restrict__ out) {
    __shared__ __bf16 s_wh[NH * TPW];              // 96 KB, [col1536][tok32]
    __shared__ __bf16 s_attn[TPW][32][32];         // 64 KB, K-padded
    __shared__ float  s_f1[TPW][32], s_f2[TPW][32];//  8 KB
    __shared__ int    s_adj[NNODE * NNODE];

    const int tid  = threadIdx.x;
    const int lane = tid & 31;
    const int wave = tid >> 5;
    const int m    = lane & 15;      // WMMA row/col within tile
    const int half = lane >> 4;      // lane half selects K group
    const int tile = blockIdx.x;     // 32-token tile index

    for (int i = tid; i < NNODE * NNODE; i += 256) s_adj[i] = adj[i];

    // ---- A fragments from x (contiguous layout: K=j+16h)
    v16bf A1[2][2];                  // [mtile][kstep]
    #pragma unroll
    for (int mt = 0; mt < 2; mt++) {
        const float* xrow = x + (size_t)(tile * TPW + mt * 16 + m) * FIN;
        #pragma unroll
        for (int ks = 0; ks < 2; ks++) {
            int b0 = ks * 32 + 16 * half;
            int b1 = ks * 32 + 16 * half + 8;
            #pragma unroll
            for (int j = 0; j < 8; j++) {
                A1[mt][ks][j]     = (__bf16)xrow[b0 + j];
                A1[mt][ks][j + 8] = (__bf16)xrow[b1 + j];
            }
        }
    }

    // ---------------- Phase 1a: Wh = x @ Wf + bp2  (32 x 1536, K=64) -----
    for (int nt = wave; nt < NT_WH; nt += 8) {
        v8f c0 = {0.f,0.f,0.f,0.f,0.f,0.f,0.f,0.f};
        v8f c1 = {0.f,0.f,0.f,0.f,0.f,0.f,0.f,0.f};
        #pragma unroll
        for (int ks = 0; ks < 2; ks++) {
            v16bf B = *(const v16bf*)(wsB + (((nt << 1) + ks) << 9) + (lane << 4));
            c0 = __builtin_amdgcn_wmma_f32_16x16x32_bf16(
                     false, A1[0][ks], false, B, (short)0, c0, false, false);
            c1 = __builtin_amdgcn_wmma_f32_16x16x32_bf16(
                     false, A1[1][ks], false, B, (short)0, c1, false, false);
        }
        float bias = wsBias[nt * 16 + m];
        // token-minor: lane's 8 rows are 8 consecutive tokens -> one b128
        int col = nt * 16 + m;
        v8bf o0, o1;
        #pragma unroll
        for (int r = 0; r < 8; r++) {
            o0[r] = (__bf16)(c0[r] + bias);
            o1[r] = (__bf16)(c1[r] + bias);
        }
        *(v8bf*)&s_wh[col * TPW + 8 * half]      = o0;   // tokens 0..15
        *(v8bf*)&s_wh[col * TPW + 16 + 8 * half] = o1;   // tokens 16..31
    }
    // ---------------- Phase 1b: [f1|f2] = x @ wf + b12  (32 x 48) --------
    if (wave < 3) {
        const int nt = NT_WH + wave;
        v8f c0 = {0.f,0.f,0.f,0.f,0.f,0.f,0.f,0.f};
        v8f c1 = {0.f,0.f,0.f,0.f,0.f,0.f,0.f,0.f};
        #pragma unroll
        for (int ks = 0; ks < 2; ks++) {
            v16bf B = *(const v16bf*)(wsB + (((nt << 1) + ks) << 9) + (lane << 4));
            c0 = __builtin_amdgcn_wmma_f32_16x16x32_bf16(
                     false, A1[0][ks], false, B, (short)0, c0, false, false);
            c1 = __builtin_amdgcn_wmma_f32_16x16x32_bf16(
                     false, A1[1][ks], false, B, (short)0, c1, false, false);
        }
        int c = wave * 16 + m;                     // 0..47
        float bias = wsB12[c];
        #pragma unroll
        for (int r = 0; r < 8; r++) {
            int tok = r + 8 * half;
            if (c < 24) {
                s_f1[tok][c]           = c0[r] + bias;
                s_f1[16 + tok][c]      = c1[r] + bias;
            } else {
                s_f2[tok][c - 24]      = c0[r] + bias;
                s_f2[16 + tok][c - 24] = c1[r] + bias;
            }
        }
    }
    __syncthreads();

    // ---------------- Phase 2: leaky-relu / mask / softmax ---------------
    for (int tt = 0; tt < 4; tt++) {
        int tk = wave * 4 + tt;
        float f2j = (lane < NNODE) ? s_f2[tk][lane] : 0.f;
        for (int i = 0; i < 32; i++) {
            __bf16 st = (__bf16)0.f;
            if (i < NNODE) {
                float e;
                if (lane < NNODE) {
                    e = s_f1[tk][i] + f2j;
                    e = e > 0.f ? e : 0.2f * e;
                    if (s_adj[i * NNODE + lane] == 0) e = -9.0e15f;
                } else e = -9.0e15f;
                float mx = e;
                for (int d = 16; d > 0; d >>= 1) mx = fmaxf(mx, __shfl_xor(mx, d, 32));
                float ex = (lane < NNODE) ? __expf(e - mx) : 0.f;
                float sm = ex;
                for (int d = 16; d > 0; d >>= 1) sm += __shfl_xor(sm, d, 32);
                float at = ex / sm;
                if (lane < NNODE) st = (__bf16)at;
            }
            s_attn[tk][i][lane] = st;   // zero for K>=24 and rows>=24
        }
    }
    __syncthreads();

    // ---------------- Phase 3: h' = attn @ Wh, elu, mean -----------------
    for (int tt = 0; tt < 4; tt++) {
        int tk = wave * 4 + tt;
        v16bf A3[2];
        #pragma unroll
        for (int mt = 0; mt < 2; mt++) {
            const __bf16* ar = &s_attn[tk][mt * 16 + m][0];
            #pragma unroll
            for (int j = 0; j < 8; j++) {
                A3[mt][j]     = ar[16 * half + j];
                A3[mt][j + 8] = ar[16 * half + 8 + j];
            }
        }
        #pragma unroll
        for (int nt = 0; nt < 4; nt++) {
            v16bf Bf;
            #pragma unroll
            for (int j = 0; j < 16; j++) {
                int k = j + 16 * half;             // B layout: lane=N, K contiguous
                Bf[j] = (k < NNODE)
                      ? s_wh[(k * HDIM + nt * 16 + m) * TPW + tk]
                      : (__bf16)0.f;
            }
            v8f d0 = {0.f,0.f,0.f,0.f,0.f,0.f,0.f,0.f};
            v8f d1 = {0.f,0.f,0.f,0.f,0.f,0.f,0.f,0.f};
            d0 = __builtin_amdgcn_wmma_f32_16x16x32_bf16(
                     false, A3[0], false, Bf, (short)0, d0, false, false);
            d1 = __builtin_amdgcn_wmma_f32_16x16x32_bf16(
                     false, A3[1], false, Bf, (short)0, d1, false, false);
            float p = 0.f;
            #pragma unroll
            for (int r = 0; r < 8; r++) {          // rows 0..15 (valid)
                float v = d0[r]; p += (v > 0.f) ? v : (__expf(v) - 1.f);
            }
            #pragma unroll
            for (int r = 0; r < 8; r++) {          // rows 16..31 (>=24 exact 0)
                float v = d1[r]; p += (v > 0.f) ? v : (__expf(v) - 1.f);
            }
            p += __shfl_xor(p, 16, 32);            // combine the two lane halves
            if (lane < 16)
                out[(size_t)(tile * TPW + tk) * HDIM + nt * 16 + lane] = p * (1.0f / 24.0f);
        }
    }
}

extern "C" void kernel_launch(void* const* d_in, const int* in_sizes, int n_in,
                              void* d_out, int out_size, void* d_ws, size_t ws_size,
                              hipStream_t stream) {
    (void)in_sizes; (void)n_in; (void)out_size; (void)ws_size;
    const float* x   = (const float*)d_in[0];
    const int*   adj = (const int*)  d_in[1];
    const float* Wp  = (const float*)d_in[2];
    const float* bp  = (const float*)d_in[3];
    const float* W   = (const float*)d_in[4];
    const float* a   = (const float*)d_in[5];
    float* out = (float*)d_out;
    __bf16* ws    = (__bf16*)d_ws;
    float* wsBias = (float*)(ws + WS_ALL_ELEMS);   // 4B-aligned (even elem count)
    float* wsB12  = wsBias + NH;

    int total = WS_ALL_ELEMS + NH + 48;
    gat_prep<<<(total + 255) / 256, 256, 0, stream>>>(Wp, bp, W, a, ws, wsBias, wsB12);
    // 32768 tokens / 32 per tile = 1024 workgroups
    gat_fused<<<1024, 256, 0, stream>>>(x, adj, ws, wsBias, wsB12, out);
}
